// PhyGraphConv_32719060861306
// MI455X (gfx1250) — matrix operand order repaired and checked by reading, not compiled, not run on hardware
//
#include <hip/hip_runtime.h>
#include <hip/hip_bf16.h>
#include <math.h>

// ---------------- problem constants ----------------
#define BB   4
#define NN   16384
#define EE   131072
#define CC   128
#define NDD  3
#define MIDD 128
#define OUTT 128
#define K1   (CC * NDD)      // 384
#define ENX  (EE + NN)       // 147456 extended edges per batch

// ---------------- workspace layout (bytes) ----------------
#define OFF_FT    ((size_t)0)                       // (B,N,C) f32         33,554,432
#define OFF_GRAD  ((size_t)33554432)                // (B,N,384) f32      100,663,296
#define OFF_FO    ((size_t)134217728)               // (B,N,128) f32       33,554,432
#define OFF_DENOM ((size_t)167772160)               // (B,N) f32              262,144
#define OFF_W1T   ((size_t)168034304)               // (128,384) bf16          98,304
#define OFF_W2T   ((size_t)168132608)               // (128,128) bf16          32,768
#define OFF_ACC   OFF_FT                            // aliases fT (dead after k_grad)

// ---------------- vector types for WMMA ----------------
typedef __attribute__((ext_vector_type(16))) __bf16 v16bf;
typedef __attribute__((ext_vector_type(8)))  __bf16 v8bf;
typedef __attribute__((ext_vector_type(4)))  __bf16 v4bf;
typedef __attribute__((ext_vector_type(8)))  float  v8f;

__device__ __forceinline__ v16bf cat8(v8bf lo, v8bf hi) {
    return __builtin_shufflevector(lo, hi, 0,1,2,3,4,5,6,7,8,9,10,11,12,13,14,15);
}

__device__ __forceinline__ void atomAddF(float* p, float v) {
    (void)__hip_atomic_fetch_add(p, v, __ATOMIC_RELAXED, __HIP_MEMORY_SCOPE_AGENT);
}

__device__ __forceinline__ float gelu_exact(float x) {
    return 0.5f * x * (1.0f + erff(x * 0.70710678118654752f));
}

// ---------------- zero scratch ----------------
__global__ void __launch_bounds__(256) k_zero(float4* p, long n4) {
    long i = (long)blockIdx.x * 256 + threadIdx.x;
    long stride = (long)gridDim.x * 256;
    float4 z = make_float4(0.f, 0.f, 0.f, 0.f);
    for (; i < n4; i += stride) p[i] = z;
}

// ---------------- transpose f (B,C,N) -> fT (B,N,C) ----------------
__global__ void __launch_bounds__(256) k_transpose(const float* __restrict__ f,
                                                   float* __restrict__ fT) {
    __shared__ float tile[32][33];
    int b  = blockIdx.z;
    int n0 = blockIdx.x * 32;
    int c0 = blockIdx.y * 32;
    int tx = threadIdx.x, ty = threadIdx.y; // (32,8)
    #pragma unroll
    for (int i = 0; i < 4; ++i)
        tile[ty + 8*i][tx] = f[((size_t)b*CC + (c0 + ty + 8*i)) * NN + n0 + tx];
    __syncthreads();
    #pragma unroll
    for (int i = 0; i < 4; ++i)
        fT[((size_t)b*NN + (n0 + ty + 8*i)) * CC + c0 + tx] = tile[tx][ty + 8*i];
}

// ---------------- convert weights to bf16 (K-major) ----------------
__global__ void __launch_bounds__(256) k_prepw(const float* __restrict__ w1,
                                               const float* __restrict__ w2,
                                               __bf16* __restrict__ w1t,
                                               __bf16* __restrict__ w2t) {
    int idx = blockIdx.x * 256 + threadIdx.x;   // 65536 total
    if (idx < MIDD * K1) {                      // w1 (128,384) -> w1t (128,384)
        int n = idx / K1, k = idx % K1;
        w1t[(size_t)n * K1 + k] = (__bf16)w1[idx];
    } else {
        int j = idx - MIDD * K1;                // w2 (128,128) -> w2t (128,128)
        int n = j / MIDD, k = j % MIDD;
        w2t[(size_t)n * MIDD + k] = (__bf16)w2[j];
    }
}

// ---------------- edge gradient scatter: one wave per edge ----------------
__global__ void __launch_bounds__(256) k_grad(const float* __restrict__ fT,
                                              const int* __restrict__ edges,
                                              const float* __restrict__ egw,
                                              float* __restrict__ grad) {
    int warp = threadIdx.x >> 5, lane = threadIdx.x & 31;
    int gid  = blockIdx.x * 8 + warp;           // < B*E = 524288
    int b = gid / EE, e = gid % EE;
    size_t eb = (size_t)b * EE + e;
    int src = edges[eb * 2 + 0];
    int tgt = edges[eb * 2 + 1];
    const float4* fs4 = (const float4*)(fT + ((size_t)b*NN + src) * CC);
    const float4* ft4 = (const float4*)(fT + ((size_t)b*NN + tgt) * CC);
    float4 a = fs4[lane];
    float4 c = ft4[lane];
    float w0 = egw[eb * 3 + 0];
    float w1 = egw[eb * 3 + 1];
    float w2 = egw[eb * 3 + 2];
    float df[4] = {a.x - c.x, a.y - c.y, a.z - c.z, a.w - c.w};
    float* gb = grad + ((size_t)b*NN + tgt) * K1 + lane * 12;  // c = 4*lane..4*lane+3
    #pragma unroll
    for (int j = 0; j < 4; ++j) {
        atomAddF(gb + j*3 + 0, df[j] * w0);
        atomAddF(gb + j*3 + 1, df[j] * w1);
        atomAddF(gb + j*3 + 2, df[j] * w2);
    }
}

// ---------------- fused MLP: WMMA GEMM1 -> GELU -> WMMA GEMM2 ----------------
// block = 256 (8 waves); tile = 32 nodes x 128 cols; wave w owns column-tile w.
__global__ void __launch_bounds__(256, 2) k_mlp(const float* __restrict__ grad,
                                                const __bf16* __restrict__ w1t,
                                                const __bf16* __restrict__ w2t,
                                                const float* __restrict__ b1,
                                                const float* __restrict__ b2,
                                                float* __restrict__ fo) {
    __shared__ __align__(16) __bf16 As[32 * 392];   // 32x384 G-tile (pitch 392)
    __shared__ __align__(16) __bf16 Hs[32 * 136];   // 32x128 H-tile (pitch 136)

    int b  = blockIdx.y;
    int n0 = blockIdx.x * 32;
    int t  = threadIdx.x;
    int w  = t >> 5, lane = t & 31;

    // ---- stage G tile (f32 -> bf16) into LDS ----
    const float* gsrc = grad + ((size_t)b*NN + n0) * K1;
    #pragma unroll
    for (int it = 0; it < 12; ++it) {
        int idx = (t + 256 * it) * 4;               // 12288 elements, float4 steps
        int r = idx / K1, k = idx % K1;
        float4 v = *(const float4*)(gsrc + (size_t)r * K1 + k);
        v4bf pk = { (__bf16)v.x, (__bf16)v.y, (__bf16)v.z, (__bf16)v.w };
        *(v4bf*)(As + r * 392 + k) = pk;
    }
    __syncthreads();

    int col    = w * 16 + (lane & 15);              // output column (C/D layout)
    int krow   = (lane & 15) + ((lane & 16) ? 16 : 0); // B-fragment K row offset
    int achunk = (lane & 16) ? 8 : 0;               // A-fragment first K chunk
    int mhi    = (lane & 16) ? 8 : 0;               // C/D row offset

    // ---- GEMM1: H(32x128) = G(32x384) @ W1^T(384x128) ----
    v8f acc0 = {}, acc1 = {};
    for (int kk = 0; kk < 12; ++kk) {
        int k0 = kk * 32;
        const __bf16* bp = w1t + (size_t)col * K1 + k0 + achunk;
        v16bf bfrag = cat8(*(const v8bf*)bp, *(const v8bf*)(bp + 16));

        const __bf16* ap0 = As + (lane & 15) * 392 + k0 + achunk;
        v16bf a0 = cat8(*(const v8bf*)ap0, *(const v8bf*)(ap0 + 16));
        acc0 = __builtin_amdgcn_wmma_f32_16x16x32_bf16(false, a0, false, bfrag,
                                                       (short)0, acc0, false, false);
        const __bf16* ap1 = As + (16 + (lane & 15)) * 392 + k0 + achunk;
        v16bf a1 = cat8(*(const v8bf*)ap1, *(const v8bf*)(ap1 + 16));
        acc1 = __builtin_amdgcn_wmma_f32_16x16x32_bf16(false, a1, false, bfrag,
                                                       (short)0, acc1, false, false);
    }

    // ---- bias + exact GELU, store H (bf16) to LDS ----
    float bias1 = b1[col];
    #pragma unroll
    for (int r = 0; r < 8; ++r) {
        Hs[(r + mhi) * 136 + col]      = (__bf16)gelu_exact(acc0[r] + bias1);
        Hs[(16 + r + mhi) * 136 + col] = (__bf16)gelu_exact(acc1[r] + bias1);
    }
    __syncthreads();

    // ---- GEMM2: FO(32x128) = H(32x128) @ W2^T(128x128) ----
    v8f o0 = {}, o1 = {};
    for (int kk = 0; kk < 4; ++kk) {
        int k0 = kk * 32;
        const __bf16* bp = w2t + (size_t)col * MIDD + k0 + achunk;
        v16bf bfrag = cat8(*(const v8bf*)bp, *(const v8bf*)(bp + 16));

        const __bf16* ap0 = Hs + (lane & 15) * 136 + k0 + achunk;
        v16bf a0 = cat8(*(const v8bf*)ap0, *(const v8bf*)(ap0 + 16));
        o0 = __builtin_amdgcn_wmma_f32_16x16x32_bf16(false, a0, false, bfrag,
                                                     (short)0, o0, false, false);
        const __bf16* ap1 = Hs + (16 + (lane & 15)) * 136 + k0 + achunk;
        v16bf a1 = cat8(*(const v8bf*)ap1, *(const v8bf*)(ap1 + 16));
        o1 = __builtin_amdgcn_wmma_f32_16x16x32_bf16(false, a1, false, bfrag,
                                                     (short)0, o1, false, false);
    }

    float bias2 = b2[col];
    float* dst = fo + ((size_t)b*NN + n0) * OUTT + col;
    #pragma unroll
    for (int r = 0; r < 8; ++r) {
        dst[(size_t)(r + mhi) * OUTT]      = o0[r] + bias2;
        dst[(size_t)(16 + r + mhi) * OUTT] = o1[r] + bias2;
    }
}

// ---------------- Gaussian edge kernel + aggregation: one wave per ext-edge ----------------
__global__ void __launch_bounds__(256) k_edge(const float* __restrict__ nodes,
                                              const float* __restrict__ nw,
                                              const int* __restrict__ edges,
                                              const float* __restrict__ Lm,
                                              const float* __restrict__ cb,
                                              const float* __restrict__ fo,
                                              float* __restrict__ acc,
                                              float* __restrict__ denom) {
    int warp = threadIdx.x >> 5, lane = threadIdx.x & 31;
    int gid  = blockIdx.x * 8 + warp;           // < B*(E+N) = 589824
    int b  = gid / ENX, ee = gid % ENX;
    int src, tgt;
    if (ee < EE) {
        size_t eb = (size_t)b * EE + ee;
        src = edges[eb * 2 + 0];
        tgt = edges[eb * 2 + 1];
    } else {
        src = tgt = ee - EE;
    }
    const float* ps = nodes + ((size_t)b*NN + src) * NDD;
    const float* pt = nodes + ((size_t)b*NN + tgt) * NDD;
    float d0 = ps[0] - pt[0], d1 = ps[1] - pt[1], d2 = ps[2] - pt[2];

    // Sigma = L @ L^T (3x3, symmetric), computed per-thread from 9 scalars
    float L00 = Lm[0], L01 = Lm[1], L02 = Lm[2];
    float L10 = Lm[3], L11 = Lm[4], L12 = Lm[5];
    float L20 = Lm[6], L21 = Lm[7], L22 = Lm[8];
    float S00 = L00*L00 + L01*L01 + L02*L02;
    float S01 = L00*L10 + L01*L11 + L02*L12;
    float S02 = L00*L20 + L01*L21 + L02*L22;
    float S11 = L10*L10 + L11*L11 + L12*L12;
    float S12 = L10*L20 + L11*L21 + L12*L22;
    float S22 = L20*L20 + L21*L21 + L22*L22;
    float dS0 = S00*d0 + S01*d1 + S02*d2 + cb[0];
    float dS1 = S01*d0 + S11*d1 + S12*d2 + cb[1];
    float dS2 = S02*d0 + S12*d1 + S22*d2 + cb[2];
    float q = d0*dS0 + d1*dS1 + d2*dS2;
    float kval = nw[(size_t)b*NN + src] * expf(-q);

    if (lane == 0) atomAddF(denom + (size_t)b*NN + tgt, kval);

    const float4* fs = (const float4*)(fo + ((size_t)b*NN + src) * OUTT);
    float4 v = fs[lane];
    float* ab = acc + ((size_t)b*NN + tgt) * OUTT + lane * 4;
    atomAddF(ab + 0, v.x * kval);
    atomAddF(ab + 1, v.y * kval);
    atomAddF(ab + 2, v.z * kval);
    atomAddF(ab + 3, v.w * kval);
}

// ---------------- normalize + transpose to (B,OUT,N) ----------------
__global__ void __launch_bounds__(256) k_final(const float* __restrict__ acc,
                                               const float* __restrict__ denom,
                                               float* __restrict__ out) {
    __shared__ float tile[32 * 129];
    __shared__ float invd[32];
    int b  = blockIdx.y;
    int n0 = blockIdx.x * 32;
    int t  = threadIdx.x;
    if (t < 32) invd[t] = 1.0f / denom[(size_t)b*NN + n0 + t];
    #pragma unroll
    for (int it = 0; it < 16; ++it) {
        int idx = t + 256 * it;                 // 4096 = 32 rows x 128 cols
        int r = idx >> 7, o = idx & 127;
        tile[r * 129 + o] = acc[((size_t)b*NN + n0 + r) * OUTT + o];
    }
    __syncthreads();
    #pragma unroll
    for (int it = 0; it < 16; ++it) {
        int idx = t + 256 * it;
        int o = idx >> 5, r = idx & 31;
        out[((size_t)b*OUTT + o) * NN + n0 + r] = tile[r * 129 + o] * invd[r];
    }
}

// ---------------- host launcher ----------------
extern "C" void kernel_launch(void* const* d_in, const int* in_sizes, int n_in,
                              void* d_out, int out_size, void* d_ws, size_t ws_size,
                              hipStream_t stream) {
    const float* f     = (const float*)d_in[0];
    const float* nodes = (const float*)d_in[1];
    const float* nw    = (const float*)d_in[2];
    const int*   edges = (const int*)d_in[3];
    const float* egw   = (const float*)d_in[4];
    const float* w1    = (const float*)d_in[5];
    const float* b1    = (const float*)d_in[6];
    const float* w2    = (const float*)d_in[7];
    const float* b2    = (const float*)d_in[8];
    const float* Lm    = (const float*)d_in[9];
    const float* cb    = (const float*)d_in[10];
    float* out = (float*)d_out;

    char* ws = (char*)d_ws;
    float*  fT    = (float*)(ws + OFF_FT);
    float*  grad  = (float*)(ws + OFF_GRAD);
    float*  fo    = (float*)(ws + OFF_FO);
    float*  denom = (float*)(ws + OFF_DENOM);
    __bf16* w1t   = (__bf16*)(ws + OFF_W1T);
    __bf16* w2t   = (__bf16*)(ws + OFF_W2T);
    float*  acc   = (float*)(ws + OFF_ACC);   // aliases fT (dead after k_grad)

    // 1. zero grad accumulator (B*N*384 floats)
    long grad4 = (long)BB * NN * K1 / 4;
    k_zero<<<4096, 256, 0, stream>>>((float4*)grad, grad4);

    // 2. transpose f -> fT
    k_transpose<<<dim3(NN/32, CC/32, BB), dim3(32, 8, 1), 0, stream>>>(f, fT);

    // 3. weights -> bf16, K-major
    k_prepw<<<(MIDD*K1 + MIDD*OUTT) / 256, 256, 0, stream>>>(w1, w2, w1t, w2t);

    // 4. edge gradient scatter (atomics into grad)
    k_grad<<<(BB * EE) / 8, 256, 0, stream>>>(fT, edges, egw, grad);

    // 5. zero acc (aliases fT, now dead) and denom
    k_zero<<<4096, 256, 0, stream>>>((float4*)acc, (long)BB * NN * OUTT / 4);
    k_zero<<<64,   256, 0, stream>>>((float4*)denom, (long)BB * NN / 4);

    // 6. fused WMMA MLP -> fo
    k_mlp<<<dim3(NN/32, BB, 1), 256, 0, stream>>>(grad, w1t, w2t, b1, b2, fo);

    // 7. Gaussian kernel aggregation (atomics into acc/denom)
    k_edge<<<(BB * ENX) / 8, 256, 0, stream>>>(nodes, nw, edges, Lm, cb, fo, acc, denom);

    // 8. normalize + transpose to (B,OUT,N)
    k_final<<<dim3(NN/32, BB, 1), 256, 0, stream>>>(acc, denom, out);
}
